// Rel1SelfAttention_21680994910232
// MI455X (gfx1250) — compile-verified, matrix operand order checked by reading, not run on hardware
//
#include <hip/hip_runtime.h>
#include <hip/hip_bf16.h>

typedef __bf16 bf16_t;
typedef __attribute__((ext_vector_type(16))) __bf16 v16bf;
typedef __attribute__((ext_vector_type(8)))  float  v8f;

static constexpr int kB   = 2;
static constexpr int kN   = 2048;
static constexpr int kHid = 768;
static constexpr int kH   = 12;
static constexpr int kD   = 64;

#define WMMA_BF16(a, b, c) \
  __builtin_amdgcn_wmma_f32_16x16x32_bf16(false, (a), false, (b), (short)0, (c), false, false)

#define S_WAIT_ASYNCCNT(n) asm volatile("s_wait_asynccnt %0" :: "i"(n) : "memory")

// Async global -> LDS copy, 16 bytes per lane (CDNA5 GLOBAL_LOAD_ASYNC_TO_LDS,
// tracked with ASYNCcnt). lds_off = LDS byte address (= low 32 bits of the
// generic shared-memory address per the flat-aperture rule).
__device__ __forceinline__ void async_b128(unsigned int lds_off, const void* gptr) {
  asm volatile("global_load_async_to_lds_b128 %0, %1, off"
               :: "v"(lds_off), "v"(gptr) : "memory");
}

// Fragment loader: lane L covers row (L&15), K-halves split by lane group.
// Used for A (16 rows x 32 K) and B^T-style (16 cols x 32 K) tiles, where the
// K dimension is contiguous in memory (32-byte v16bf loads -> b128 pairs).
__device__ __forceinline__ v16bf load_frag(const bf16_t* base, int stride, int lane) {
  return *(const v16bf*)(base + (size_t)(lane & 15) * stride + (size_t)(lane >> 4) * 16);
}

// ---------------------------------------------------------------------------
// Kernel 1: fp32 -> bf16 staging (+ r_emb transpose (N,H,D) -> (H,N,D))
// ---------------------------------------------------------------------------
__global__ void convert_kernel(const float* __restrict__ hs,
                               const float* __restrict__ Wq,
                               const float* __restrict__ Wk,
                               const float* __restrict__ Wv,
                               const float* __restrict__ remb,
                               bf16_t* __restrict__ hsb,
                               bf16_t* __restrict__ wb,
                               bf16_t* __restrict__ reb) {
  long gid = (long)blockIdx.x * blockDim.x + threadIdx.x;
  long gsz = (long)gridDim.x * blockDim.x;

  const long nHS = (long)kB * kN * kHid;
  for (long i = gid; i < nHS; i += gsz) hsb[i] = (bf16_t)hs[i];

  const long nW = (long)kHid * kHid;
  for (long i = gid; i < nW; i += gsz) {
    wb[i]          = (bf16_t)Wq[i];
    wb[nW + i]     = (bf16_t)Wk[i];
    wb[2 * nW + i] = (bf16_t)Wv[i];
  }

  const long nRE = (long)kH * kN * kD;
  for (long i = gid; i < nRE; i += gsz) {
    int h   = (int)(i / ((long)kN * kD));
    long r  = i % ((long)kN * kD);
    int j   = (int)(r / kD);
    int d   = (int)(r % kD);
    reb[i] = (bf16_t)remb[((long)j * kH + h) * kD + d];
  }
}

// ---------------------------------------------------------------------------
// Kernel 2: QKV projection, bf16 WMMA GEMM, 128x128 block tile, K-step 32.
// Double-buffered GLOBAL_LOAD_ASYNC_TO_LDS pipeline (ASYNCcnt), so global
// latency overlaps WMMA issue and no VGPRs are burned on staging.
//   out[m,o] = sum_k hs[m,k] * W[o,k] + bias[o]   (NT gemm)
// Epilogue scatters into attention-friendly layouts:
//   qw/qp/k : (B,H,N,D)   v : (B,H,D,N)  (transposed for contiguous PV frags)
// ---------------------------------------------------------------------------
__global__ __launch_bounds__(256) void qkv_gemm_kernel(
    const bf16_t* __restrict__ hsb, const bf16_t* __restrict__ wb,
    const float* __restrict__ bq, const float* __restrict__ bk,
    const float* __restrict__ bv, const float* __restrict__ rwb,
    bf16_t* __restrict__ qw, bf16_t* __restrict__ qp,
    bf16_t* __restrict__ kbo, bf16_t* __restrict__ vtb) {
  __shared__ bf16_t At[2][128 * 32];
  __shared__ bf16_t Bt[2][128 * 32];

  const int tid  = threadIdx.x;
  const int lane = tid & 31;
  const int wave = tid >> 5;          // 8 waves (wave32)
  const int wr   = wave >> 2;         // 0..1  (64-row strips)
  const int wc   = wave & 3;          // 0..3  (32-col strips)
  const int m0   = blockIdx.x * 128;
  const int o0   = blockIdx.y * 128;
  const int z    = blockIdx.z;        // 0=q 1=k 2=v
  const bf16_t* wmat = wb + (size_t)z * kHid * kHid;

  v8f vzero = {};
  v8f acc[4][2];
#pragma unroll
  for (int a = 0; a < 4; ++a)
#pragma unroll
    for (int c = 0; c < 2; ++c) acc[a][c] = vzero;

  const int lrow = tid >> 1;          // 0..127
  const int lseg = (tid & 1) * 16;    // 0 / 16 (halfs)

  const bf16_t* gArow = hsb  + (size_t)(m0 + lrow) * kHid + lseg;
  const bf16_t* gBrow = wmat + (size_t)(o0 + lrow) * kHid + lseg;
  unsigned int ldsA[2], ldsB[2];
  ldsA[0] = (unsigned int)(uintptr_t)&At[0][lrow * 32 + lseg];
  ldsA[1] = (unsigned int)(uintptr_t)&At[1][lrow * 32 + lseg];
  ldsB[0] = (unsigned int)(uintptr_t)&Bt[0][lrow * 32 + lseg];
  ldsB[1] = (unsigned int)(uintptr_t)&Bt[1][lrow * 32 + lseg];

  // Prologue: async-stage tile 0.
  async_b128(ldsA[0],      gArow);
  async_b128(ldsA[0] + 16, gArow + 8);
  async_b128(ldsB[0],      gBrow);
  async_b128(ldsB[0] + 16, gBrow + 8);

  const int KT = kHid / 32;           // 24 K-steps
  for (int kt = 0; kt < KT; ++kt) {
    const int buf = kt & 1;
    if (kt + 1 < KT) {                // prefetch tile kt+1 into other buffer
      const bf16_t* ga = gArow + (size_t)(kt + 1) * 32;
      const bf16_t* gb = gBrow + (size_t)(kt + 1) * 32;
      async_b128(ldsA[buf ^ 1],      ga);
      async_b128(ldsA[buf ^ 1] + 16, ga + 8);
      async_b128(ldsB[buf ^ 1],      gb);
      async_b128(ldsB[buf ^ 1] + 16, gb + 8);
      S_WAIT_ASYNCCNT(4);             // tile kt landed; kt+1 still in flight
    } else {
      S_WAIT_ASYNCCNT(0);
    }
    __syncthreads();

    const bf16_t* Ab = At[buf];
    const bf16_t* Bb = Bt[buf];
    v16bf bfrag[2];
#pragma unroll
    for (int nt = 0; nt < 2; ++nt)
      bfrag[nt] = load_frag(&Bb[(wc * 32 + nt * 16) * 32], 32, lane);
#pragma unroll
    for (int mt = 0; mt < 4; ++mt) {
      v16bf afrag = load_frag(&Ab[(wr * 64 + mt * 16) * 32], 32, lane);
#pragma unroll
      for (int nt = 0; nt < 2; ++nt)
        acc[mt][nt] = WMMA_BF16(afrag, bfrag[nt], acc[mt][nt]);
    }
    __syncthreads();                  // reads done before buf is re-filled
  }

  const float* bias = (z == 0) ? bq : (z == 1) ? bk : bv;
  const int rbase = 8 * (lane >> 4);
  const int ncol  = lane & 15;
#pragma unroll
  for (int mt = 0; mt < 4; ++mt)
#pragma unroll
    for (int nt = 0; nt < 2; ++nt) {
      v8f a = acc[mt][nt];
#pragma unroll
      for (int i = 0; i < 8; ++i) {
        int m  = m0 + wr * 64 + mt * 16 + rbase + i;
        int o  = o0 + wc * 32 + nt * 16 + ncol;
        int bb = m >> 11;          // /kN
        int np = m & (kN - 1);
        int hh = o >> 6;           // /kD
        int dd = o & (kD - 1);
        float val = a[i] + bias[o];
        size_t idx = (((size_t)bb * kH + hh) * kN + np) * kD + dd;
        if (z == 0) {
          qp[idx] = (bf16_t)val;
          qw[idx] = (bf16_t)(val + rwb[o]);
        } else if (z == 1) {
          kbo[idx] = (bf16_t)val;
        } else {
          vtb[(((size_t)bb * kH + hh) * kD + dd) * kN + np] = (bf16_t)val;
        }
      }
    }
}

// ---------------------------------------------------------------------------
// Kernel 3: fused relative attention, one block per (b, h, 16 query rows).
//   Phase 1: BD strip  BDraw[r][j] = qp[p0+r]·re[j] + r_bias[j]  (rows 0..16)
//   Phase 2: flash loop over 32-key tiles; rel_shift resolved by LDS gather:
//       q<=p : BDraw[p][N+q-p-1],  q==p+1 : 0,  q>p+1 : BDraw[p+1][q-p-2]
// ---------------------------------------------------------------------------
__global__ __launch_bounds__(128) void rel_attn_kernel(
    const bf16_t* __restrict__ qw, const bf16_t* __restrict__ qp,
    const bf16_t* __restrict__ kb, const bf16_t* __restrict__ vtb,
    const bf16_t* __restrict__ reb, const float* __restrict__ r_bias,
    float* __restrict__ out) {
  extern __shared__ char smem[];
  bf16_t* strip  = (bf16_t*)smem;                                     // 17*kN bf16
  float*  combO  = (float*)(smem + (size_t)17 * kN * sizeof(bf16_t)); // 4*16*64
  float*  combM  = combO + 4 * 16 * 64;                               // 4*16
  float*  combL  = combM + 4 * 16;                                    // 4*16
  bf16_t* pstage = (bf16_t*)(combL + 4 * 16);                         // 4*16*32

  const int tid  = threadIdx.x;
  const int lane = tid & 31;
  const int wave = tid >> 5;          // 4 waves
  const int p0   = blockIdx.x * 16;
  const int h    = blockIdx.y;
  const int b    = blockIdx.z;
  const size_t bh = (size_t)b * kH + h;
  const int rbase = 8 * (lane >> 4);
  const int ncol  = lane & 15;

  // ---------------- Phase 1: BD strip via WMMA ----------------
  const bf16_t* qpb  = qp + (bh * kN + p0) * kD;
  const bf16_t* rebh = reb + (size_t)h * kN * kD;
  v16bf ap0 = load_frag(qpb + 0,  kD, lane);
  v16bf ap1 = load_frag(qpb + 32, kD, lane);

  for (int jc = wave; jc < kN / 32; jc += 4) {
    int j0 = jc * 32;
#pragma unroll
    for (int nh2 = 0; nh2 < 2; ++nh2) {
      int jb = j0 + nh2 * 16;
      v16bf b0 = load_frag(rebh + (size_t)jb * kD + 0,  kD, lane);
      v16bf b1 = load_frag(rebh + (size_t)jb * kD + 32, kD, lane);
      v8f acc = {};
      acc = WMMA_BF16(ap0, b0, acc);
      acc = WMMA_BF16(ap1, b1, acc);
      int j = jb + ncol;
      float rbj = r_bias[(size_t)j * kH + h];
#pragma unroll
      for (int i = 0; i < 8; ++i)
        strip[(size_t)(rbase + i) * kN + j] = (bf16_t)(acc[i] + rbj);
    }
  }
  // Extra row p0+16 (needed by rel_shift upper region): plain VALU dots.
  if (p0 + 16 < kN) {
    const bf16_t* q16 = qp + (bh * kN + p0 + 16) * kD;
    for (int j = tid; j < kN; j += 128) {
      const bf16_t* rr = rebh + (size_t)j * kD;
      float s = 0.f;
#pragma unroll
      for (int d2 = 0; d2 < kD; ++d2) s += (float)q16[d2] * (float)rr[d2];
      strip[(size_t)16 * kN + j] = (bf16_t)(s + r_bias[(size_t)j * kH + h]);
    }
  }
  __syncthreads();

  // ---------------- Phase 2: flash attention ----------------
  const bf16_t* qwb   = qw + (bh * kN + p0) * kD;
  const bf16_t* kbase = kb + bh * kN * kD;
  const bf16_t* vbase = vtb + bh * (size_t)kD * kN;
  v16bf aw0 = load_frag(qwb + 0,  kD, lane);
  v16bf aw1 = load_frag(qwb + 32, kD, lane);

  v8f vzero = {};
  v8f oacc[4];
#pragma unroll
  for (int t = 0; t < 4; ++t) oacc[t] = vzero;
  float m_run[8], l_run[8];
#pragma unroll
  for (int i = 0; i < 8; ++i) { m_run[i] = -1e30f; l_run[i] = 0.f; }

  bf16_t* pw = pstage + wave * (16 * 32);

  for (int jt = wave; jt < kN / 32; jt += 4) {
    int j0 = jt * 32;
    if (jt + 4 < kN / 32) {           // GL2 prefetch of this wave's next tiles
      __builtin_prefetch(kbase + (size_t)(j0 + 128) * kD, 0, 1);
      __builtin_prefetch(vbase + (size_t)(j0 + 128), 0, 1);
    }
    float sc[2][8];
#pragma unroll
    for (int nh2 = 0; nh2 < 2; ++nh2) {
      int jb = j0 + nh2 * 16;
      v16bf k0 = load_frag(kbase + (size_t)jb * kD + 0,  kD, lane);
      v16bf k1 = load_frag(kbase + (size_t)jb * kD + 32, kD, lane);
      v8f acc = vzero;
      acc = WMMA_BF16(aw0, k0, acc);
      acc = WMMA_BF16(aw1, k1, acc);
      int q = jb + ncol;
#pragma unroll
      for (int i = 0; i < 8; ++i) {
        int p = p0 + rbase + i;
        float bd;
        if (q <= p)
          bd = (float)strip[(size_t)(rbase + i) * kN + (kN + q - p - 1)];
        else if (q == p + 1)
          bd = 0.f;
        else
          bd = (float)strip[(size_t)(rbase + i + 1) * kN + (q - p - 2)];
        sc[nh2][i] = (acc[i] + bd) * 0.125f;   // 1/sqrt(64)
      }
    }
    // Online softmax (row reductions within 16-lane halves).
#pragma unroll
    for (int i = 0; i < 8; ++i) {
      float tmax = fmaxf(sc[0][i], sc[1][i]);
      tmax = fmaxf(tmax, __shfl_xor(tmax, 1, 32));
      tmax = fmaxf(tmax, __shfl_xor(tmax, 2, 32));
      tmax = fmaxf(tmax, __shfl_xor(tmax, 4, 32));
      tmax = fmaxf(tmax, __shfl_xor(tmax, 8, 32));
      float mn   = fmaxf(m_run[i], tmax);
      float corr = __expf(m_run[i] - mn);
      float p0e  = __expf(sc[0][i] - mn);
      float p1e  = __expf(sc[1][i] - mn);
      float ps   = p0e + p1e;
      ps += __shfl_xor(ps, 1, 32);
      ps += __shfl_xor(ps, 2, 32);
      ps += __shfl_xor(ps, 4, 32);
      ps += __shfl_xor(ps, 8, 32);
      l_run[i] = l_run[i] * corr + ps;
      m_run[i] = mn;
#pragma unroll
      for (int t2 = 0; t2 < 4; ++t2) oacc[t2][i] *= corr;
      pw[(rbase + i) * 32 + ncol]      = (bf16_t)p0e;
      pw[(rbase + i) * 32 + 16 + ncol] = (bf16_t)p1e;
    }
    asm volatile("s_wait_dscnt 0" ::: "memory");  // P stage visible to own wave
    v16bf pa = load_frag(pw, 32, lane);
#pragma unroll
    for (int t2 = 0; t2 < 4; ++t2) {
      v16bf vf = load_frag(vbase + (size_t)(t2 * 16) * kN + j0, kN, lane);
      oacc[t2] = WMMA_BF16(pa, vf, oacc[t2]);
    }
  }

  // ---------------- split-K combine across 4 waves ----------------
  if (ncol == 0) {
#pragma unroll
    for (int i = 0; i < 8; ++i) {
      combM[wave * 16 + rbase + i] = m_run[i];
      combL[wave * 16 + rbase + i] = l_run[i];
    }
  }
#pragma unroll
  for (int t2 = 0; t2 < 4; ++t2)
#pragma unroll
    for (int i = 0; i < 8; ++i)
      combO[wave * 1024 + (rbase + i) * 64 + t2 * 16 + ncol] = oacc[t2][i];
  __syncthreads();

  for (int e = tid; e < 16 * 64; e += 128) {
    int r = e >> 6, d = e & 63;
    float M = -1e30f;
#pragma unroll
    for (int w = 0; w < 4; ++w) M = fmaxf(M, combM[w * 16 + r]);
    float L = 0.f, O = 0.f;
#pragma unroll
    for (int w = 0; w < 4; ++w) {
      float f = __expf(combM[w * 16 + r] - M);
      L += combL[w * 16 + r] * f;
      O += combO[w * 1024 + r * 64 + d] * f;
    }
    out[((size_t)b * kN + p0 + r) * kHid + (size_t)h * kD + d] = O / L;
  }
}

// ---------------------------------------------------------------------------
extern "C" void kernel_launch(void* const* d_in, const int* in_sizes, int n_in,
                              void* d_out, int out_size, void* d_ws, size_t ws_size,
                              hipStream_t stream) {
  (void)in_sizes; (void)n_in; (void)out_size; (void)ws_size;
  const float* hs   = (const float*)d_in[0];
  const float* remb = (const float*)d_in[1];
  const float* rwb  = (const float*)d_in[2];
  const float* rb   = (const float*)d_in[3];
  const float* Wq   = (const float*)d_in[4];
  const float* bq   = (const float*)d_in[5];
  const float* Wk   = (const float*)d_in[6];
  const float* bk   = (const float*)d_in[7];
  const float* Wv   = (const float*)d_in[8];
  const float* bv   = (const float*)d_in[9];
  float* out = (float*)d_out;

  bf16_t* w = (bf16_t*)d_ws;
  size_t off = 0;
  bf16_t* hsb = w + off; off += (size_t)kB * kN * kHid;
  bf16_t* wb  = w + off; off += (size_t)3 * kHid * kHid;
  bf16_t* reb = w + off; off += (size_t)kH * kN * kD;
  bf16_t* qwp = w + off; off += (size_t)kB * kH * kN * kD;
  bf16_t* qpp = w + off; off += (size_t)kB * kH * kN * kD;
  bf16_t* kbp = w + off; off += (size_t)kB * kH * kN * kD;
  bf16_t* vtp = w + off; off += (size_t)kB * kH * kN * kD;

  convert_kernel<<<1024, 256, 0, stream>>>(hs, Wq, Wk, Wv, remb, hsb, wb, reb);
  qkv_gemm_kernel<<<dim3(kB * kN / 128, kHid / 128, 3), 256, 0, stream>>>(
      hsb, wb, bq, bk, bv, rwb, qwp, qpp, kbp, vtp);

  size_t smem = (size_t)17 * kN * sizeof(bf16_t) +
                (size_t)(4 * 16 * 64 + 4 * 16 + 4 * 16) * sizeof(float) +
                (size_t)4 * 16 * 32 * sizeof(bf16_t);
  rel_attn_kernel<<<dim3(kN / 16, kH, kB), 128, smem, stream>>>(
      qwp, qpp, kbp, vtp, reb, rb, out);
}
